// RNNBase_52802327937082
// MI455X (gfx1250) — compile-verified, matrix operand order checked by reading, not run on hardware
//
#include <hip/hip_runtime.h>
#include <hip/hip_bf16.h>
#include <stdint.h>

#define T_DIM 512
#define B_DIM 64
#define H_DIM 1024
#define L_DIM 2
#define LDS_STRIDE 516           // dwords per staged row: 16B-aligned, conflict-friendly
#define HBF_STRIDE (H_DIM / 2)   // u32 per row of packed-bf16 h

// Packed-weight geometry (uint32 units): lane stride 8, ks stride 32*8=256,
// column-tile stride 32*32*8=8192 (=> 1KB / 32KB byte strides: fit IOFFSET).
#define PK_LANE   8
#define PK_KS     256
#define PK_CT     8192

#if defined(__has_builtin)
#if __has_builtin(__builtin_amdgcn_sched_group_barrier)
#define HAVE_SGB 1
#endif
#endif
// sched_group_barrier masks: 0x008 = MFMA/WMMA, 0x020 = VMEM read, 0x100 = DS read
#ifdef HAVE_SGB
#define SGB(mask, size) __builtin_amdgcn_sched_group_barrier((mask), (size), 0)
#else
#define SGB(mask, size) do {} while (0)
#endif

typedef __attribute__((ext_vector_type(16))) __bf16 v16bf;
typedef __attribute__((ext_vector_type(8)))  float  v8f;

union Frag16 { uint32_t u[8]; uint4 q[2]; v16bf v; };
union AccF   { float f[8]; v8f v; };

__device__ __forceinline__ uint32_t pack2_bf16(float a, float b) {
  uint32_t ua = __float_as_uint(a), ub = __float_as_uint(b);
  ua = (ua + 0x7FFFu + ((ua >> 16) & 1u)) >> 16;   // RNE f32->bf16
  ub = (ub + 0x7FFFu + ((ub >> 16) & 1u)) >> 16;
  return ua | (ub << 16);
}

// Pack one HxH f32 weight matrix (row c = output feature, col k) into
// WMMA B-fragment order: dst[((ct*32 + ks)*32 + lane)*8 + v] holds the
// bf16 pair for (n = ct*16 + lane%16, k = ks*32 + kfrag(v, lane/16)).
__global__ void __launch_bounds__(256)
pack_weight_kernel(const float* __restrict__ W, uint32_t* __restrict__ dst) {
  const int tid  = blockIdx.x * blockDim.x + threadIdx.x;
  const int lane = tid & 31;
  const int ks   = (tid >> 5) & 31;
  const int ct   = tid >> 10;
  const int c    = ct * 16 + (lane & 15);
  const int half = lane >> 4;
  Frag16 o;
#pragma unroll
  for (int v = 0; v < 8; ++v) {
    const int kf = (v < 4 ? 2 * v : 16 + 2 * (v - 4)) + 8 * half;
    const int k  = ks * 32 + kf;
    o.u[v] = pack2_bf16(W[(size_t)c * H_DIM + k], W[(size_t)c * H_DIM + k + 1]);
  }
  uint4* dp = (uint4*)(dst + ((size_t)tid << 3));
  dp[0] = o.q[0];
  dp[1] = o.q[1];
}

__global__ void bias_sum_kernel(const float* __restrict__ a,
                                const float* __restrict__ b,
                                float* __restrict__ o, int n) {
  int i = blockIdx.x * blockDim.x + threadIdx.x;
  if (i < n) o[i] = a[i] + b[i];
}

__global__ void copy_kernel(const float* __restrict__ s, float* __restrict__ d, int n) {
  int i = blockIdx.x * blockDim.x + threadIdx.x;
  if (i < n) d[i] = s[i];
}

// f32 row-major -> packed bf16 k-pairs (u32 per pair), n = #pairs.
__global__ void pack_pairs_kernel(const float* __restrict__ s,
                                  uint32_t* __restrict__ d, int n) {
  int i = blockIdx.x * blockDim.x + threadIdx.x;
  if (i < n) {
    const float2 v = ((const float2*)s)[i];
    d[i] = pack2_bf16(v.x, v.y);
  }
}

// xw = in @ W^T + bias over (T*B) rows. Each block owns 16 rows x all 1024
// cols; rows are staged to LDS BEFORE any store, so in==out aliasing
// (layer 1) is safe. 8 waves x 8 column tiles, set-level double buffering,
// with a sched-group pattern pinning 2-loads-per-WMMA interleave so VMEM
// overlaps the matrix pipe instead of serializing on loadcnt 0.
__global__ void __launch_bounds__(256)
gemm_xw_kernel(const float* in, float* out,
               const uint32_t* __restrict__ wpk, const float* __restrict__ bias) {
  __shared__ uint32_t lds[16 * LDS_STRIDE];
  const int row0 = blockIdx.x * 16;
  const int tid  = threadIdx.x;

#pragma unroll
  for (int i = 0; i < 16; ++i) {
    const int p = tid + i * 256;        // 0..4095 quads
    const int r = p >> 8;
    const int q = p & 255;
    const float4 v = ((const float4*)(in + (size_t)(row0 + r) * H_DIM))[q];
    uint2 w;
    w.x = pack2_bf16(v.x, v.y);
    w.y = pack2_bf16(v.z, v.w);
    *(uint2*)&lds[r * LDS_STRIDE + q * 2] = w;
  }
  __syncthreads();

  const int lane = tid & 31;
  const int wave = tid >> 5;
  const int m    = lane & 15;
  const int half = lane >> 4;

  AccF acc[8];
#pragma unroll
  for (int i = 0; i < 8; ++i) {
    const float bz = bias[(wave * 8 + i) * 16 + m];
#pragma unroll
    for (int r = 0; r < 8; ++r) acc[i].f[r] = bz;
  }

  const uint32_t* bbase = wpk + (size_t)wave * 8 * PK_CT + lane * PK_LANE;
  const uint32_t* arow  = &lds[m * LDS_STRIDE + 4 * half];

  Frag16 a[2];
  Frag16 b[2][8];
#define LOADSET(ks, s)                                                        \
  do {                                                                        \
    a[s].q[0] = *(const uint4*)(arow + (ks) * 16);                            \
    a[s].q[1] = *(const uint4*)(arow + (ks) * 16 + 8);                        \
    _Pragma("unroll")                                                         \
    for (int _i = 0; _i < 8; ++_i) {                                          \
      b[s][_i].q[0] = *(const uint4*)(bbase + (ks) * PK_KS + _i * PK_CT);     \
      b[s][_i].q[1] = *(const uint4*)(bbase + (ks) * PK_KS + _i * PK_CT + 4); \
    }                                                                         \
  } while (0)

  LOADSET(0, 0);
#pragma unroll 2
  for (int ks = 0; ks < 32; ++ks) {
    const int cur = ks & 1;
    if (ks + 1 < 32) LOADSET(ks + 1, cur ^ 1);
#pragma unroll
    for (int i = 0; i < 8; ++i)
      acc[i].v = __builtin_amdgcn_wmma_f32_16x16x32_bf16(
          false, a[cur].v, false, b[cur][i].v, (short)0, acc[i].v, false, false);
    // Issue-steering: A-frag DS reads first, then interleave the next set's
    // B loads two-at-a-time with this set's 8 WMMAs.
    SGB(0x100, 2);
#pragma unroll
    for (int i = 0; i < 8; ++i) {
      SGB(0x020, 2);
      SGB(0x008, 1);
    }
  }
#undef LOADSET

#pragma unroll
  for (int i = 0; i < 8; ++i) {
    const int n = (wave * 8 + i) * 16 + m;
    float* op = out + (size_t)(row0 + half * 8) * H_DIM + n;
#pragma unroll
    for (int r = 0; r < 8; ++r) op[(size_t)r * H_DIM] = acc[i].f[r];
  }
}

// One recurrent step: h_new = tanh(xw_t + h_prev @ Whh^T).
// No LDS, no barrier: A fragments load straight from the packed-bf16 h
// ping buffer (128 KB, L2-resident) with two b128 loads per k-step.
// Fully unrolled serial chain with prefetch distance 3 (4 slots); the
// sched-group pattern (4 VMEM reads : 1 WMMA) keeps loads issued 3 steps
// ahead so the waits stay partial. Outputs: ys (f32, in place over xw)
// and h packed bf16 for the next step (even lanes pack (n, n+1) via shfl).
__global__ void __launch_bounds__(256)
rnn_step_kernel(const uint32_t* __restrict__ hbf_prev,
                uint32_t* __restrict__ hbf_next,
                float* ys, const uint32_t* __restrict__ wpk) {
  const int tid  = threadIdx.x;
  const int lane = tid & 31;
  const int wave = tid >> 5;
  const int m    = lane & 15;
  const int half = lane >> 4;
  const int row0 = blockIdx.x * 16;           // batch rows
  const int ct   = blockIdx.y * 8 + wave;     // column tile
  const int n    = ct * 16 + m;

  AccF acc;
  float* yp = ys + (size_t)(row0 + half * 8) * H_DIM + n;
#pragma unroll
  for (int r = 0; r < 8; ++r) acc.f[r] = yp[(size_t)r * H_DIM];

  const uint32_t* abase = hbf_prev + (size_t)(row0 + m) * HBF_STRIDE + 4 * half;
  const uint32_t* bbase = wpk + (size_t)ct * PK_CT + lane * PK_LANE;

  Frag16 a[4], b[4];
#define LOAD_FRAG(ks, s)                                            \
  do {                                                              \
    a[s].q[0] = *(const uint4*)(abase + (ks) * 16);                 \
    a[s].q[1] = *(const uint4*)(abase + (ks) * 16 + 8);             \
    b[s].q[0] = *(const uint4*)(bbase + (ks) * PK_KS);              \
    b[s].q[1] = *(const uint4*)(bbase + (ks) * PK_KS + 4);          \
  } while (0)

  LOAD_FRAG(0, 0);
  LOAD_FRAG(1, 1);
  LOAD_FRAG(2, 2);
#pragma unroll
  for (int ks = 0; ks < 32; ++ks) {
    if (ks + 3 < 32) LOAD_FRAG(ks + 3, (ks + 3) & 3);
    acc.v = __builtin_amdgcn_wmma_f32_16x16x32_bf16(
        false, a[ks & 3].v, false, b[ks & 3].v, (short)0, acc.v, false, false);
    SGB(0x020, 4);   // the 4 prefetch loads for step ks+3
    SGB(0x008, 1);   // this step's WMMA
  }
#undef LOAD_FRAG

  uint32_t* hnp = hbf_next + (size_t)(row0 + half * 8) * HBF_STRIDE
                + (ct * 8 + (m >> 1));
#pragma unroll
  for (int r = 0; r < 8; ++r) {
    const float v  = tanhf(acc.f[r]);
    const float vx = __shfl_xor(v, 1);        // odd neighbor's value
    yp[(size_t)r * H_DIM] = v;
    if ((m & 1) == 0) hnp[(size_t)r * HBF_STRIDE] = pack2_bf16(v, vx);
  }
}

extern "C" void kernel_launch(void* const* d_in, const int* in_sizes, int n_in,
                              void* d_out, int out_size, void* d_ws, size_t ws_size,
                              hipStream_t stream) {
  (void)in_sizes; (void)n_in; (void)out_size; (void)ws_size;

  const float* x    = (const float*)d_in[0];  // (T,B,H)
  const float* h0   = (const float*)d_in[1];  // (L,B,H)
  const float* W_ih = (const float*)d_in[2];  // (L,H,H)
  const float* W_hh = (const float*)d_in[3];  // (L,H,H)
  const float* b_ih = (const float*)d_in[4];  // (L,H)
  const float* b_hh = (const float*)d_in[5];  // (L,H)

  float* out    = (float*)d_out;
  float* outcur = out;                                   // ys / xw region (T,B,H)
  float* outhid = out + (size_t)T_DIM * B_DIM * H_DIM;   // hidden (L,B,H)

  // Workspace: packed bf16 weights (8 MB) + fused bias + 2 packed-h buffers.
  const size_t pk_elems = (size_t)H_DIM * H_DIM / 2;     // u32 per matrix per layer
  uint32_t* wih_pk = (uint32_t*)d_ws;
  uint32_t* whh_pk = wih_pk + (size_t)L_DIM * pk_elems;
  float*    bias   = (float*)(whh_pk + (size_t)L_DIM * pk_elems);
  uint32_t* hbf0   = (uint32_t*)(bias + (size_t)L_DIM * H_DIM);
  uint32_t* hbf1   = hbf0 + (size_t)B_DIM * HBF_STRIDE;

  const int packBlocks = (64 * 32 * 32) / 256;
  for (int l = 0; l < L_DIM; ++l) {
    pack_weight_kernel<<<packBlocks, 256, 0, stream>>>(
        W_ih + (size_t)l * H_DIM * H_DIM, wih_pk + (size_t)l * pk_elems);
    pack_weight_kernel<<<packBlocks, 256, 0, stream>>>(
        W_hh + (size_t)l * H_DIM * H_DIM, whh_pk + (size_t)l * pk_elems);
  }
  bias_sum_kernel<<<(L_DIM * H_DIM) / 256, 256, 0, stream>>>(
      b_ih, b_hh, bias, L_DIM * H_DIM);

  const int hPairs = B_DIM * H_DIM / 2;   // 32768
  for (int l = 0; l < L_DIM; ++l) {
    const float* in = (l == 0) ? x : outcur;   // layer 1 reads ys0 in place (safe)
    gemm_xw_kernel<<<(T_DIM * B_DIM) / 16, 256, 0, stream>>>(
        in, outcur, wih_pk + (size_t)l * pk_elems, bias + (size_t)l * H_DIM);

    // Seed packed h with h0 for this layer.
    pack_pairs_kernel<<<hPairs / 256, 256, 0, stream>>>(
        h0 + (size_t)l * B_DIM * H_DIM, hbf0, hPairs);

    for (int t = 0; t < T_DIM; ++t) {
      const uint32_t* hp = (t & 1) ? hbf1 : hbf0;
      uint32_t*       hn = (t & 1) ? hbf0 : hbf1;
      rnn_step_kernel<<<dim3(B_DIM / 16, H_DIM / 128), 256, 0, stream>>>(
          hp, hn, outcur + (size_t)t * B_DIM * H_DIM,
          whh_pk + (size_t)l * pk_elems);
    }

    // hidden[l] = h_T = ys[T-1] (must be saved before next layer overwrites).
    copy_kernel<<<(B_DIM * H_DIM) / 256, 256, 0, stream>>>(
        outcur + (size_t)(T_DIM - 1) * B_DIM * H_DIM,
        outhid + (size_t)l * B_DIM * H_DIM, B_DIM * H_DIM);
  }
}
